// NeuralODE_90958817395362
// MI455X (gfx1250) — matrix operand order repaired and checked by reading, not compiled, not run on hardware
//
#include <hip/hip_runtime.h>
#include <hip/hip_bf16.h>
#include <math.h>

// Problem dims (match reference: B=512, D=H=1024, T=33)
#define MDIM 512
#define KDIM 1024
#define NDIM 1024

typedef __attribute__((ext_vector_type(16))) __bf16 v16bf;
typedef __attribute__((ext_vector_type(8)))  __bf16 v8bf;
typedef __attribute__((ext_vector_type(8)))  float  v8f;
typedef __attribute__((ext_vector_type(4)))  unsigned int u32x4;
typedef __attribute__((ext_vector_type(8)))  int i32x8;
typedef __attribute__((ext_vector_type(4)))  int i32x4;

static __device__ __forceinline__ v16bf cat8(v8bf lo, v8bf hi) {
  return __builtin_shufflevector(lo, hi, 0,1,2,3,4,5,6,7,8,9,10,11,12,13,14,15);
}

// ---------------------------------------------------------------------------
// TDM: DMA a [rows x 32] bf16 tile (row stride 1024 elems) from global into
// LDS, with LDS padding 4 DWORDs after every 16 DWORDs of data -> row pitch
// 40 halves (80 B), matching the compute-side fragment addressing.
// D# layout per CDNA5 ISA.
// clang-23 toolchain: 6-arg builtin (g0, g1, g2, g3, extra, cpol).
// ---------------------------------------------------------------------------
static __device__ __forceinline__ void tdm_load_tile(unsigned lds_byte_off,
                                                     const __bf16* gptr,
                                                     int rows) {
  unsigned long long ga = (unsigned long long)(size_t)gptr;
  u32x4 g0;
  g0[0] = 1u;                                            // count=1 (valid), user mode
  g0[1] = lds_byte_off;                                  // lds_addr [63:32]
  g0[2] = (unsigned)(ga & 0xFFFFFFFFu);                  // global_addr [95:64]
  g0[3] = (unsigned)((ga >> 32) & 0x01FFFFFFu)           // global_addr [120:96]
        | (2u << 30);                                    // type=2 ("image") [127:126]
  i32x8 g1;
  g1[0] = (int)((1u << 16)      // data_size = 1 -> 2 bytes
              | (1u << 20)      // pad_enable
              | (3u << 22)      // pad_interval = 3 -> 16 DWORDs of data
              | (3u << 25));    // pad_amount   = 3 -> 4 DWORDs of pad
  g1[1] = (int)(1024u << 16);   // tensor_dim0[15:0] at bits 63:48
  g1[2] = (int)(1024u << 16);   // tensor_dim0 hi = 0 ; tensor_dim1[15:0]
  g1[3] = (int)(32u << 16);     // tensor_dim1 hi = 0 ; tile_dim0 = 32
  g1[4] = rows;                 // tile_dim1 = rows ; tile_dim2 = 0 (unused)
  g1[5] = 1024;                 // tensor_dim0_stride[31:0] = 1024 elems
  g1[6] = 0;                    // tensor_dim0_stride hi ; tensor_dim1_stride lo
  g1[7] = 0;
  i32x4 gz4 = {0, 0, 0, 0};             // groups 2/3 unused (2D tile)
  i32x8 gz8 = {0, 0, 0, 0, 0, 0, 0, 0}; // extra group (6-arg form)
  __builtin_amdgcn_tensor_load_to_lds(g0, g1, gz4, gz4, gz8, 0);
}

// ---------------------------------------------------------------------------
// One-time prep: W (f32, [K][N]) -> WT (bf16, [N][K]) so every WMMA fragment
// load is K-contiguous (lane = N, two runs of 8 contiguous K halves).
// ---------------------------------------------------------------------------
__global__ void prep_weights(const float* __restrict__ W1,
                             const float* __restrict__ W2,
                             __bf16* __restrict__ W1T,
                             __bf16* __restrict__ W2T) {
  int idx = blockIdx.x * 256 + threadIdx.x;
  int n = idx >> 10;
  int k = idx & (KDIM - 1);
  W1T[idx] = (__bf16)W1[(size_t)k * NDIM + n];
  W2T[idx] = (__bf16)W2[(size_t)k * NDIM + n];
}

__global__ void copy_z(const float* __restrict__ src, float* __restrict__ dst) {
  int idx = blockIdx.x * 256 + threadIdx.x;
  dst[idx] = src[idx];
}

// ---------------------------------------------------------------------------
// Fused GEMM for one RK4 stage.
//   IS_G1 == 1 :  h = tanh((z + c_s*k_prev) @ W1 + t_s*wt + b1)
//                 (A built f32->bf16 by the waves; B via TDM)
//   IS_G1 == 0 :  k_s = h @ W2 + b2 ; fused RK4 z-accumulation
//                 (A and B both via TDM)
// Block: 256 threads = 8 waves; block tile 64(M) x 128(N); wave tile 32x32 =
// 2x2 v_wmma_f32_16x16x32_bf16; K-chunk 32; double-buffered LDS; wave 0 runs
// the Tensor Data Mover, all waves sync on one barrier per K-chunk.
// ---------------------------------------------------------------------------
template<int IS_G1>
__global__ __launch_bounds__(256) void odegemm(
    const float*  __restrict__ Zin,    // G1: z base (f32)
    const float*  __restrict__ Kprev,  // G1: previous k (f32)  (== z for stage 1)
    const __bf16* __restrict__ Abf,    // G2: h (bf16)
    const __bf16* __restrict__ WT,     // transposed bf16 weights [N][K]
    const float*  __restrict__ v0,     // G1: wt ; G2: b2
    const float*  __restrict__ v1,     // G1: b1 ; G2: unused
    const float*  __restrict__ tarr,   // time array (device)
    int step, int stage,
    __bf16* __restrict__ Hout,         // G1 output (bf16 activations)
    float*  __restrict__ Zbuf,         // G2: z (read stage1, written stage4)
    float*  __restrict__ Kout,         // G2: k_s output (stages 1..3)
    float*  __restrict__ Zacc)         // G2: RK4 accumulator
{
  const float t0 = tarr[step];
  const float dt = tarr[step + 1] - t0;

  const int tid  = threadIdx.x;
  const int lane = tid & 31;
  const int wm   = (tid >> 5) >> 2;   // 0..1 : wave row in block tile
  const int wn   = (tid >> 5) & 3;    // 0..3 : wave col in block tile
  const int m0   = blockIdx.y * 64;
  const int n0   = blockIdx.x * 128;
  const bool dma = (tid < 32);        // wave 0 drives the TDM

  __shared__ __align__(16) __bf16 As[2][64][40];   // 40-half pitch = TDM pad layout
  __shared__ __align__(16) __bf16 Bs[2][128][40];

  // manual A staging (GEMM1 only): 64 rows x 32 K, 8 f32 per thread
  const int ar = tid >> 2, ac = (tid & 3) * 8;

  float coef = 0.f, ts = t0;
  if (IS_G1) {
    if (stage == 2 || stage == 3) { coef = 0.5f * dt; ts = t0 + 0.5f * dt; }
    else if (stage == 4)          { coef = dt;        ts = t0 + dt; }
  }
  const float wk = dt * (1.0f / 6.0f) * ((stage == 2 || stage == 3) ? 2.0f : 1.0f);

  v8f acc[2][2];
  #pragma unroll
  for (int a = 0; a < 2; ++a)
    #pragma unroll
    for (int b = 0; b < 2; ++b)
      #pragma unroll
      for (int r = 0; r < 8; ++r) acc[a][b][r] = 0.0f;

  float a_f[8];

  auto issue_tdm = [&](int buf, int kb) {
    tdm_load_tile((unsigned)(size_t)&Bs[buf][0][0],
                  WT + (size_t)n0 * KDIM + kb, 128);
    if (!IS_G1)
      tdm_load_tile((unsigned)(size_t)&As[buf][0][0],
                    Abf + (size_t)m0 * KDIM + kb, 64);
  };

  auto load_a = [&](int kb) {
    if (IS_G1) {
      const float* zp = Zin   + (size_t)(m0 + ar) * KDIM + kb + ac;
      const float* kp = Kprev + (size_t)(m0 + ar) * KDIM + kb + ac;
      float4 z0 = *(const float4*)zp, z1 = *(const float4*)(zp + 4);
      float4 p0 = *(const float4*)kp, p1 = *(const float4*)(kp + 4);
      a_f[0] = z0.x + coef * p0.x;  a_f[1] = z0.y + coef * p0.y;
      a_f[2] = z0.z + coef * p0.z;  a_f[3] = z0.w + coef * p0.w;
      a_f[4] = z1.x + coef * p1.x;  a_f[5] = z1.y + coef * p1.y;
      a_f[6] = z1.z + coef * p1.z;  a_f[7] = z1.w + coef * p1.w;
      if (kb + 64 < KDIM) __builtin_prefetch(zp + 64, 0, 0);   // global_prefetch
    }
  };

  auto store_a = [&](int buf) {
    if (IS_G1) {
      v8bf av;
      #pragma unroll
      for (int j = 0; j < 8; ++j) av[j] = (__bf16)a_f[j];
      *(v8bf*)&As[buf][ar][ac] = av;
    }
  };

  auto compute = [&](int buf) {
    // A 16x32 bf16 layout: lane l -> M = l&15, K = (l>>4)*8..+8 and +16..+24
    const int arow = wm * 32 + (lane & 15);
    const int acol = (lane >> 4) * 8;
    // B 32x16 layout: lane l -> N = l&15, K = (l>>4)*8..+8 and +16..+24
    const int brow = wn * 32 + (lane & 15);
    const int bcol = (lane >> 4) * 8;
    v16bf af[2], bq[2];
    #pragma unroll
    for (int a = 0; a < 2; ++a) {
      const __bf16* p = &As[buf][arow + a * 16][0];
      af[a] = cat8(*(const v8bf*)(p + acol), *(const v8bf*)(p + acol + 16));
    }
    #pragma unroll
    for (int b = 0; b < 2; ++b) {
      const __bf16* p = &Bs[buf][brow + b * 16][0];
      bq[b] = cat8(*(const v8bf*)(p + bcol), *(const v8bf*)(p + bcol + 16));
    }
    #pragma unroll
    for (int a = 0; a < 2; ++a)
      #pragma unroll
      for (int b = 0; b < 2; ++b)
        acc[a][b] = __builtin_amdgcn_wmma_f32_16x16x32_bf16(
            false, af[a], false, bq[b], (short)0, acc[a][b], false, false);
  };

  // ---- prologue: DMA + stage chunk 0 ----
  if (dma) issue_tdm(0, 0);
  load_a(0);
  store_a(0);
  if (dma) __builtin_amdgcn_s_wait_tensorcnt(0);
  __syncthreads();

  // ---- main loop: double-buffered, one barrier per K-chunk ----
  const int NIT = KDIM / 32;
  for (int it = 0; it < NIT; ++it) {
    const int cur = it & 1;
    const bool more = (it + 1) < NIT;
    if (more) {
      if (dma) issue_tdm(cur ^ 1, (it + 1) * 32);   // DMA next tiles
      load_a((it + 1) * 32);                        // G1: fused z + c*k -> regs
    }
    compute(cur);
    if (more) store_a(cur ^ 1);
    if (more && dma) __builtin_amdgcn_s_wait_tensorcnt(0);
    __syncthreads();
  }

  // ---- epilogue. C/D layout: VGPR r, lane l -> M = (l>>4)*8 + r, N = l&15 ----
  #pragma unroll
  for (int a = 0; a < 2; ++a) {
    #pragma unroll
    for (int b = 0; b < 2; ++b) {
      const int col  = n0 + wn * 32 + b * 16 + (lane & 15);
      const int rowb = m0 + wm * 32 + a * 16 + (lane >> 4) * 8;
      if (IS_G1) {
        const float wtn = v0[col], b1n = v1[col];
        #pragma unroll
        for (int r = 0; r < 8; ++r) {
          float h = tanhf(acc[a][b][r] + ts * wtn + b1n);
          Hout[(size_t)(rowb + r) * NDIM + col] = (__bf16)h;
        }
      } else {
        const float b2n = v0[col];
        #pragma unroll
        for (int r = 0; r < 8; ++r) {
          const size_t idx = (size_t)(rowb + r) * NDIM + col;
          const float kv = acc[a][b][r] + b2n;
          if (stage == 1)       { Kout[idx] = kv; Zacc[idx] = Zbuf[idx] + wk * kv; }
          else if (stage < 4)   { Kout[idx] = kv; Zacc[idx] = Zacc[idx] + wk * kv; }
          else                  { Zbuf[idx] = Zacc[idx] + wk * kv; }
        }
      }
    }
  }
}

// ---------------------------------------------------------------------------
extern "C" void kernel_launch(void* const* d_in, const int* in_sizes, int n_in,
                              void* d_out, int out_size, void* d_ws, size_t ws_size,
                              hipStream_t stream) {
  const float* z_init = (const float*)d_in[0];
  const float* t      = (const float*)d_in[1];
  const float* W1     = (const float*)d_in[2];
  const float* b1     = (const float*)d_in[3];
  const float* wt     = (const float*)d_in[4];
  const float* W2     = (const float*)d_in[5];
  const float* b2     = (const float*)d_in[6];

  float* z = (float*)d_out;                         // z lives in d_out (B x D f32)
  char*  ws = (char*)d_ws;
  __bf16* W1T = (__bf16*)(ws);                      // 2 MB
  __bf16* W2T = (__bf16*)(ws + (size_t)(2u << 20));
  __bf16* Hbf = (__bf16*)(ws + (size_t)(4u << 20)); // 1 MB activations
  float*  Kbf = (float*)(ws + (size_t)(5u << 20));  // 2 MB k_s
  float*  Zac = (float*)(ws + (size_t)(7u << 20));  // 2 MB accumulator

  prep_weights<<<(NDIM * KDIM) / 256, 256, 0, stream>>>(W1, W2, W1T, W2T);
  copy_z<<<(MDIM * KDIM) / 256, 256, 0, stream>>>(z_init, z);

  const dim3 grid(NDIM / 128, MDIM / 64);   // 8 x 8 = 64 workgroups
  const int n_steps = in_sizes[1] - 1;      // T-1 = 32 RK4 steps

  for (int s = 0; s < n_steps; ++s) {
    for (int st = 1; st <= 4; ++st) {
      odegemm<1><<<grid, 256, 0, stream>>>(
          z, (st == 1 ? z : Kbf), nullptr, W1T, wt, b1, t, s, st,
          Hbf, nullptr, nullptr, nullptr);
      odegemm<0><<<grid, 256, 0, stream>>>(
          nullptr, nullptr, Hbf, W2T, b2, nullptr, t, s, st,
          nullptr, z, Kbf, Zac);
    }
  }
}
